// Attention_25786983645664
// MI455X (gfx1250) — compile-verified, matrix operand order checked by reading, not run on hardware
//
#include <hip/hip_runtime.h>
#include <hip/hip_bf16.h>

// ---------------------------------------------------------------------------
// CDNA5 (gfx1250) wave32 WMMA attention pipeline.
//   HEADS=16, DIM_HEAD=128, b=2, n=2048, dim=2048, inner=2048
// ---------------------------------------------------------------------------

typedef __attribute__((ext_vector_type(16))) __bf16 v16bf;
typedef __attribute__((ext_vector_type(8)))  float  v8f;
typedef __attribute__((may_alias)) uint4 uint4_a;
typedef __attribute__((ext_vector_type(4))) unsigned int u32x4;
typedef __attribute__((ext_vector_type(8))) int i32x8;
typedef __attribute__((ext_vector_type(4))) int i32x4;

union V8F   { v8f v; float f[8]; };
union BF16x16 { v16bf v; uint4 u[2]; __bf16 e[16]; };

#define N_TOK   2048
#define DIM     2048
#define HEADS   16
#define DHEAD   128
#define INNER   2048
#define BATCH   2
#define M_TOT   (BATCH * N_TOK)          // 4096 rows
#define QKV_N   (3 * INNER)              // 6144
#define SCALE_Q 0.08838834764831845f     // 128^-0.5
#define NEG_BIG (-3.0e38f)

static __device__ __forceinline__ v8f wmma_bf16(v16bf a, v16bf b, v8f c) {
  // D = A(16x32 bf16) * B(32x16 bf16) + C(16x16 f32)
  return __builtin_amdgcn_wmma_f32_16x16x32_bf16(
      /*neg_a=*/false, a, /*neg_b=*/false, b,
      /*c_mod=*/(short)0, c, /*reuse_a=*/false, /*reuse_b=*/false);
}

// Load one A/B operand slice for a K=32 group from a row-major row pointer.
// Per ISA 7.12.2 (16-bit A 16x32): lane's data = bytes [k0+8*half, +16B) and
// [k0+16+8*half, +16B) packed into elements 0..7 / 8..15.
static __device__ __forceinline__ v16bf load_op(const __bf16* __restrict__ row,
                                                int k0, int half) {
  BF16x16 r;
  r.u[0] = *(const uint4_a*)(row + k0 + half * 8);
  r.u[1] = *(const uint4_a*)(row + k0 + 16 + half * 8);
  return r.v;
}

// Tensor Data Mover: DMA a 32x128 bf16 tile (row stride 128 elems) from
// global memory into LDS. Descriptor per CDNA5 ISA
// (D# group0: count=1, lds_addr, 57-bit global addr, type=2 "image";
//  group1: data_size=2B, tensor_dim0=128, tensor_dim1=32, tile_dim0=128,
//  tile_dim1=32, tensor_dim0_stride=128). Groups 2/3 zero (2D tile).
// Issued by one wave; EXEC is ignored for tensor ops; tracked by TENSORcnt.
// This toolchain exposes the 6-arg builtin form:
//   (u32x4 g0, i32x8 g1, i32x4 g2, i32x4 g3, i32x8, i32 cpol)
static __device__ __forceinline__ void tdm_load_tile_32x128(
    const __bf16* gsrc, unsigned lds_off) {
  const unsigned long long ga = (unsigned long long)(uintptr_t)gsrc;
  u32x4 g0;
  g0.x = 1u;                                                  // count=1
  g0.y = lds_off;                                             // lds_addr (bytes)
  g0.z = (unsigned)(ga & 0xFFFFFFFFu);                        // global_addr[31:0]
  g0.w = (unsigned)((ga >> 32) & 0x01FFFFFFu) | 0x80000000u;  // [56:32] | type=2
  i32x8 g1;
  g1[0] = 0x00010000;         // workgroup_mask=0, data_size=1 (2 bytes)
  g1[1] = (int)(128u << 16);  // tensor_dim0[15:0] = 128 (bits 79:48 of group1)
  g1[2] = (int)(32u << 16);   // tensor_dim0[31:16]=0 | tensor_dim1[15:0] = 32
  g1[3] = (int)(128u << 16);  // tensor_dim1[31:16]=0 | tile_dim0 = 128
  g1[4] = 32;                 // tile_dim1 = 32, tile_dim2 = 0
  g1[5] = 128;                // tensor_dim0_stride[31:0] = 128 elems (256 B)
  g1[6] = 0;                  // stride0[47:32]=0 | tensor_dim1_stride lo = 0
  g1[7] = 0;
  const i32x4 z4 = {0, 0, 0, 0};
  const i32x8 z8 = {0, 0, 0, 0, 0, 0, 0, 0};
  __builtin_amdgcn_tensor_load_to_lds(g0, g1, z4, z4, z8, /*cpol=*/0);
}

// ---------------------------------------------------------------------------
// 1) f32 -> bf16 conversion
// ---------------------------------------------------------------------------
__global__ void cvt_bf16_kernel(const float* __restrict__ src,
                                __bf16* __restrict__ dst, int n) {
  int i = blockIdx.x * blockDim.x + threadIdx.x;
  if (i < n) dst[i] = (__bf16)src[i];
}

// ---------------------------------------------------------------------------
// 2) GEMM: C[M][N] = A[M][K] * B[N][K]^T  (A,B bf16 row-major; C f32)
//    8 waves/WG, each wave a 32x32 output tile, 256 rows per WG in M.
// ---------------------------------------------------------------------------
__global__ __launch_bounds__(256) void gemm_bf16_kernel(
    const __bf16* __restrict__ A, const __bf16* __restrict__ B,
    float* __restrict__ C, int M, int N, int K) {
  (void)M;
  const int lane = threadIdx.x & 31;
  const int wave = threadIdx.x >> 5;
  const int half = lane >> 4;
  const int lm   = lane & 15;
  const int n0 = blockIdx.x * 32;
  const int m0 = blockIdx.y * 256 + wave * 32;

  const __bf16* a0 = A + (size_t)(m0 + lm) * K;
  const __bf16* a1 = A + (size_t)(m0 + 16 + lm) * K;
  const __bf16* b0 = B + (size_t)(n0 + lm) * K;
  const __bf16* b1 = B + (size_t)(n0 + 16 + lm) * K;

  V8F acc[2][2];
#pragma unroll
  for (int i = 0; i < 2; ++i)
#pragma unroll
    for (int j = 0; j < 2; ++j) acc[i][j].v = (v8f){};

  for (int k0 = 0; k0 < K; k0 += 32) {
    __builtin_prefetch(a0 + k0 + 256, 0, 0);   // -> global_prefetch_b8
    __builtin_prefetch(b0 + k0 + 256, 0, 0);
    v16bf va0 = load_op(a0, k0, half);
    v16bf va1 = load_op(a1, k0, half);
    v16bf vb0 = load_op(b0, k0, half);
    v16bf vb1 = load_op(b1, k0, half);
    acc[0][0].v = wmma_bf16(va0, vb0, acc[0][0].v);
    acc[0][1].v = wmma_bf16(va0, vb1, acc[0][1].v);
    acc[1][0].v = wmma_bf16(va1, vb0, acc[1][0].v);
    acc[1][1].v = wmma_bf16(va1, vb1, acc[1][1].v);
  }

#pragma unroll
  for (int i = 0; i < 2; ++i)
#pragma unroll
    for (int j = 0; j < 2; ++j)
#pragma unroll
      for (int v = 0; v < 8; ++v) {
        const int m = m0 + i * 16 + v + 8 * half;  // C layout: row = v + 8*(lane/16)
        const int n = n0 + j * 16 + lm;            //           col = lane%16
        C[(size_t)m * N + n] = acc[i][j].f[v];
      }
}

// ---------------------------------------------------------------------------
// 3) RoPE: qkv_raw f32 (4096 x 6144) -> q/k/v bf16 in (b,h,n,d); q scaled.
// ---------------------------------------------------------------------------
__global__ void rope_kernel(const float* __restrict__ qkv,
                            const float* __restrict__ rope,
                            __bf16* __restrict__ Qo, __bf16* __restrict__ Ko,
                            __bf16* __restrict__ Vo) {
  const int idx = blockIdx.x * blockDim.x + threadIdx.x;   // 4096*2048 threads
  if (idx >= M_TOT * INNER) return;
  const int t  = idx >> 11;          // token (b*2048 + n)
  const int hd = idx & 2047;
  const int h  = hd >> 7, d = hd & 127;
  const int n  = t & (N_TOK - 1), b = t >> 11;

  const float f = rope[n * DHEAD + d];
  const float c = __cosf(f), s = __sinf(f);
  const int d2 = (d < 64) ? d + 64 : d - 64;

  const size_t base  = (size_t)t * QKV_N + h * DHEAD;
  const size_t obase = ((size_t)(b * HEADS + h) * N_TOK + n) * DHEAD + d;

  { // q (scale commutes with the rotation)
    const float v0 = qkv[base + d], vp = qkv[base + d2];
    const float rot = (d < 64) ? -vp : vp;
    Qo[obase] = (__bf16)(SCALE_Q * (v0 * c + rot * s));
  }
  { // k
    const float v0 = qkv[base + INNER + d], vp = qkv[base + INNER + d2];
    const float rot = (d < 64) ? -vp : vp;
    Ko[obase] = (__bf16)(v0 * c + rot * s);
  }
  { // v (reference applies rotary to v as well)
    const float v0 = qkv[base + 2 * INNER + d], vp = qkv[base + 2 * INNER + d2];
    const float rot = (d < 64) ? -vp : vp;
    Vo[obase] = (__bf16)(v0 * c + rot * s);
  }
}

// ---------------------------------------------------------------------------
// 4) Causal flash attention. Grid (n/128, b*h); 256 threads = 8 waves.
//    Wave w owns query rows [q0 + 16w, +16), holds Q in A-layout registers.
//    Per 32-key tile: K tile DMA'd into LDS by the Tensor Data Mover
//    (wave 0 issues tensor_load_to_lds, drains TENSORcnt, workgroup barrier
//    releases the other waves); V staged transposed via plain loads (TDM
//    cannot transpose). S via 4x2 WMMA, online softmax in C-layout
//    (shfl_xor trees), P round-tripped through per-wave LDS into A-layout,
//    O += P*V via 8 WMMA. Output bf16 as (b, n, h*128+d).
// ---------------------------------------------------------------------------
__global__ __launch_bounds__(256) void flash_attn_kernel(
    const __bf16* __restrict__ Q, const __bf16* __restrict__ Km,
    const __bf16* __restrict__ Vm, __bf16* __restrict__ O) {
  __shared__ __bf16 ldsK[32 * 128];    // key tile, row-major (key, d) - TDM dest
  __shared__ __bf16 ldsVt[128 * 32];   // value tile, transposed (d, key)
  __shared__ __bf16 ldsP[8][16 * 32];  // per-wave P staging (row-major 16x32)

  const int lane = threadIdx.x & 31;
  const int wave = threadIdx.x >> 5;
  const int half = lane >> 4;
  const int lm   = lane & 15;

  const int bh = blockIdx.y;
  const int b = bh >> 4, h = bh & 15;
  const int q0 = blockIdx.x * 128;

  const __bf16* Qb = Q  + (size_t)bh * N_TOK * DHEAD;
  const __bf16* Kb = Km + (size_t)bh * N_TOK * DHEAD;
  const __bf16* Vb = Vm + (size_t)bh * N_TOK * DHEAD;

  const unsigned ldsK_off = (unsigned)(uintptr_t)(void*)&ldsK[0];

  // Q rows for this wave in A-operand registers (4 K-groups of 32).
  const int qrow = q0 + wave * 16 + lm;
  BF16x16 qa[4];
#pragma unroll
  for (int g = 0; g < 4; ++g) {
    const __bf16* p = Qb + (size_t)qrow * DHEAD + g * 32;
    qa[g].u[0] = *(const uint4_a*)(p + half * 8);
    qa[g].u[1] = *(const uint4_a*)(p + 16 + half * 8);
  }

  V8F acc[8];
  float mrow[8], lrow[8];
#pragma unroll
  for (int g = 0; g < 8; ++g) acc[g].v = (v8f){};
#pragma unroll
  for (int v = 0; v < 8; ++v) { mrow[v] = NEG_BIG; lrow[v] = 0.0f; }

  const int ktiles = q0 / 32 + 4;  // keys 0 .. q0+127 (causal bound for WG)
  for (int kt = 0; kt < ktiles; ++kt) {
    const int j0 = kt * 32;

    // Async stage: K tile 32x128 bf16 (8 KB) via Tensor Data Mover.
    // Safe: end-of-loop barrier guarantees no wave still reads ldsK.
    if (wave == 0) {
      tdm_load_tile_32x128(Kb + (size_t)j0 * DHEAD, ldsK_off);
    }
    // Cooperative stage: V tile transposed to (d, key).
    {
      const int j  = threadIdx.x >> 3;
      const int d0 = (threadIdx.x & 7) * 16;
      BF16x16 tv;
      const __bf16* vp = Vb + (size_t)(j0 + j) * DHEAD + d0;
      tv.u[0] = *(const uint4_a*)vp;
      tv.u[1] = *(const uint4_a*)(vp + 8);
#pragma unroll
      for (int ii = 0; ii < 16; ++ii) ldsVt[(d0 + ii) * 32 + j] = tv.e[ii];
    }
    if (wave == 0) {
      __builtin_amdgcn_s_wait_tensorcnt(0);  // K tile landed in LDS
    }
    __syncthreads();

    // S = Q * K^T for two 16-key column halves.
    V8F s0, s1;
    s0.v = (v8f){}; s1.v = (v8f){};
#pragma unroll
    for (int g = 0; g < 4; ++g) {
      BF16x16 bk;
      const __bf16* kp0 = ldsK + (size_t)lm * DHEAD + g * 32;
      bk.u[0] = *(const uint4_a*)(kp0 + half * 8);
      bk.u[1] = *(const uint4_a*)(kp0 + 16 + half * 8);
      s0.v = wmma_bf16(qa[g].v, bk.v, s0.v);
      const __bf16* kp1 = ldsK + (size_t)(16 + lm) * DHEAD + g * 32;
      bk.u[0] = *(const uint4_a*)(kp1 + half * 8);
      bk.u[1] = *(const uint4_a*)(kp1 + 16 + half * 8);
      s1.v = wmma_bf16(qa[g].v, bk.v, s1.v);
    }

    // Causal mask + online softmax update (C layout: row = v + 8*half).
    const int jg0 = j0 + lm, jg1 = j0 + 16 + lm;
#pragma unroll
    for (int v = 0; v < 8; ++v) {
      const int mq = q0 + wave * 16 + v + 8 * half;
      if (jg0 > mq) s0.f[v] = NEG_BIG;
      if (jg1 > mq) s1.f[v] = NEG_BIG;
      float t = fmaxf(s0.f[v], s1.f[v]);
      t = fmaxf(t, __shfl_xor(t, 1));
      t = fmaxf(t, __shfl_xor(t, 2));
      t = fmaxf(t, __shfl_xor(t, 4));
      t = fmaxf(t, __shfl_xor(t, 8));
      const float mnew = fmaxf(mrow[v], t);
      const float sc = __expf(mrow[v] - mnew);
      mrow[v] = mnew;
      const float p0 = __expf(s0.f[v] - mnew);
      const float p1 = __expf(s1.f[v] - mnew);
      float ps = p0 + p1;
      ps += __shfl_xor(ps, 1);
      ps += __shfl_xor(ps, 2);
      ps += __shfl_xor(ps, 4);
      ps += __shfl_xor(ps, 8);
      lrow[v] = lrow[v] * sc + ps;
#pragma unroll
      for (int g = 0; g < 8; ++g) acc[g].f[v] *= sc;
      // Stage P (bf16) row-major so it can be re-read in A layout.
      const int prow = v + 8 * half;
      ldsP[wave][prow * 32 + lm]      = (__bf16)p0;
      ldsP[wave][prow * 32 + 16 + lm] = (__bf16)p1;
    }

    // O += P * V  (A = P 16x32, B = Vt columns = head dims).
    BF16x16 pa;
    const __bf16* pp = &ldsP[wave][lm * 32];
    pa.u[0] = *(const uint4_a*)(pp + half * 8);
    pa.u[1] = *(const uint4_a*)(pp + 16 + half * 8);
#pragma unroll
    for (int g = 0; g < 8; ++g) {
      BF16x16 bv;
      const __bf16* vtp = ldsVt + (g * 16 + lm) * 32;
      bv.u[0] = *(const uint4_a*)(vtp + half * 8);
      bv.u[1] = *(const uint4_a*)(vtp + 16 + half * 8);
      acc[g].v = wmma_bf16(pa.v, bv.v, acc[g].v);
    }
    __syncthreads();  // protect ldsK/ldsVt before next tile's staging
  }

  // Normalize and emit (b, n, h*128 + d) bf16 for the projection GEMM.
#pragma unroll
  for (int v = 0; v < 8; ++v) {
    const float inv = 1.0f / lrow[v];
    const int tok = q0 + wave * 16 + v + 8 * half;
    __bf16* orow = O + ((size_t)(b * N_TOK + tok)) * INNER + h * DHEAD;
#pragma unroll
    for (int g = 0; g < 8; ++g) orow[g * 16 + lm] = (__bf16)(acc[g].f[v] * inv);
  }
}

// ---------------------------------------------------------------------------
// 5) Row-wise mean/var normalization with gain g (population variance).
// ---------------------------------------------------------------------------
__global__ __launch_bounds__(256) void layernorm_kernel(
    const float* __restrict__ X, const float* __restrict__ g,
    float* __restrict__ out) {
  __shared__ float r1[256], r2[256];
  const int row = blockIdx.x;
  const float* x = X + (size_t)row * DIM;
  float s = 0.0f, s2 = 0.0f;
  for (int c = threadIdx.x; c < DIM; c += 256) {
    const float v = x[c];
    s += v; s2 += v * v;
  }
  r1[threadIdx.x] = s; r2[threadIdx.x] = s2;
  __syncthreads();
  for (int off = 128; off > 0; off >>= 1) {
    if ((int)threadIdx.x < off) {
      r1[threadIdx.x] += r1[threadIdx.x + off];
      r2[threadIdx.x] += r2[threadIdx.x + off];
    }
    __syncthreads();
  }
  const float mean = r1[0] * (1.0f / DIM);
  const float var  = r2[0] * (1.0f / DIM) - mean * mean;
  const float rs = rsqrtf(var + 1e-5f);
  for (int c = threadIdx.x; c < DIM; c += 256)
    out[(size_t)row * DIM + c] = (x[c] - mean) * rs * g[c];
}

// ---------------------------------------------------------------------------
// Host launcher
// ---------------------------------------------------------------------------
extern "C" void kernel_launch(void* const* d_in, const int* in_sizes, int n_in,
                              void* d_out, int out_size, void* d_ws,
                              size_t ws_size, hipStream_t stream) {
  (void)in_sizes; (void)n_in; (void)out_size; (void)ws_size;
  const float* x     = (const float*)d_in[0];
  // d_in[1] = mask: all-true in this problem setup (only causal mask matters)
  const float* rope  = (const float*)d_in[2];
  const float* w_qkv = (const float*)d_in[3];
  const float* w_out = (const float*)d_in[4];
  const float* g     = (const float*)d_in[5];
  float* out = (float*)d_out;

  char* ws = (char*)d_ws;
  size_t off = 0;
  auto carve = [&](size_t bytes) -> void* {
    void* p = ws + off;
    off += (bytes + 255) & ~(size_t)255;
    return p;
  };

  const size_t NX   = (size_t)M_TOT * DIM;     // 8.4M
  const size_t NWQ  = (size_t)QKV_N * DIM;     // 12.6M
  const size_t NWO  = (size_t)DIM * INNER;     // 4.2M
  const size_t NQKV = (size_t)M_TOT * QKV_N;   // 25.2M

  __bf16* x_bf    = (__bf16*)carve(NX * 2);    // also reused as attn output
  __bf16* wqkv_bf = (__bf16*)carve(NWQ * 2);
  __bf16* wout_bf = (__bf16*)carve(NWO * 2);
  float*  qkv_raw = (float*)carve(NQKV * 4);   // also reused as proj output
  __bf16* q_bf    = (__bf16*)carve(NX * 2);
  __bf16* k_bf    = (__bf16*)carve(NX * 2);
  __bf16* v_bf    = (__bf16*)carve(NX * 2);
  __bf16* attn_bf = x_bf;                      // x consumed by QKV GEMM first
  float*  proj    = qkv_raw;                   // qkv_raw consumed by RoPE first

  cvt_bf16_kernel<<<(int)((NX  + 255) / 256), 256, 0, stream>>>(x, x_bf, (int)NX);
  cvt_bf16_kernel<<<(int)((NWQ + 255) / 256), 256, 0, stream>>>(w_qkv, wqkv_bf, (int)NWQ);
  cvt_bf16_kernel<<<(int)((NWO + 255) / 256), 256, 0, stream>>>(w_out, wout_bf, (int)NWO);

  gemm_bf16_kernel<<<dim3(QKV_N / 32, M_TOT / 256), 256, 0, stream>>>(
      x_bf, wqkv_bf, qkv_raw, M_TOT, QKV_N, DIM);

  rope_kernel<<<(int)(NX / 256), 256, 0, stream>>>(qkv_raw, rope, q_bf, k_bf, v_bf);

  flash_attn_kernel<<<dim3(N_TOK / 128, BATCH * HEADS), 256, 0, stream>>>(
      q_bf, k_bf, v_bf, attn_bf);

  gemm_bf16_kernel<<<dim3(DIM / 32, M_TOT / 256), 256, 0, stream>>>(
      attn_bf, wout_bf, proj, M_TOT, DIM, INNER);

  layernorm_kernel<<<M_TOT, 256, 0, stream>>>(proj, g, out);
}
